// TemporalGAT_30365418783391
// MI455X (gfx1250) — matrix operand order repaired and checked by reading, not compiled, not run on hardware
//
#include <hip/hip_runtime.h>
#include <hip/hip_bf16.h>
#include <math.h>

// ---------------------------------------------------------------------------
// 3-layer GAT (PyG GATConv semantics) for MI455X / gfx1250, wave32.
// GEMMs: v_wmma_f32_16x16x32_f16, one A-fragment reused across 4 B-subtiles
// (wave computes 16x64 of output). W tile staged in LDS pre-packed in the
// B-fragment register layout so each fragment is two ds_load_b128.
// Edge softmax/aggregation: 3-pass segment softmax with f32 atomics; ~170MB
// working set in d_ws so gather/scatter resolves in the 192MB L2.
// ---------------------------------------------------------------------------

#define NNODES 50000
#define NEDGES 800000
#define ETOT   (NEDGES + NNODES)   // self-loops appended
#define INDIM  128
#define HID    64
#define HEADS  4
#define GEMM_WAVES 5               // 160 threads; 3125 M-tiles = 625 blocks exactly
#define KPSTR  20                  // LDS column stride in dwords (16 kpairs + pad)

typedef __attribute__((ext_vector_type(16))) _Float16 v16h;
typedef __attribute__((ext_vector_type(8)))  float    v8f;

// ---- float atomic max via sign-split int/uint atomics (order-preserving) ----
__device__ __forceinline__ void atomicMaxF(float* addr, float val) {
  if (val >= 0.0f) atomicMax((int*)addr, __float_as_int(val));
  else             atomicMin((unsigned int*)addr, __float_as_uint(val));
}

__global__ void fill_kernel(float* __restrict__ p, float v, int n) {
  int i = blockIdx.x * blockDim.x + threadIdx.x;
  if (i < n) p[i] = v;
}

// ---------------------------------------------------------------------------
// GEMM: Y[M, Nout] = X[M, K] * W[K, Nout], fp32 in/out, f16 WMMA compute.
// Each wave: 16 rows x 64 cols (4 accumulators, shared A fragment).
// Block: 5 waves -> 80 rows x 64 cols. K multiple of 32, Nout multiple of 64.
// ---------------------------------------------------------------------------
__global__ void __launch_bounds__(GEMM_WAVES * 32)
gemm_wmma(const float* __restrict__ X, const float* __restrict__ W,
          float* __restrict__ Y, int K, int Nout) {
  // sW2[col][kpair]: one dword = f16 pair (K=2p, K=2p+1) of W[k0+2p][col0+col].
  // Matches b-operand VGPR layout directly -> fragment = 2x ds_load_b128.
  __shared__ unsigned int sW2[64 * KPSTR];

  const int lane  = threadIdx.x & 31;
  const int wave  = threadIdx.x >> 5;
  const int mtile = blockIdx.x * GEMM_WAVES + wave;
  const int col0  = blockIdx.y * 64;
  const bool hi   = lane >= 16;
  const int  lm   = lane & 15;

  const float* xp = X + (mtile * 16 + lm) * K;   // < 12.8M, 32-bit safe

  v8f acc0 = {0.f,0.f,0.f,0.f,0.f,0.f,0.f,0.f};
  v8f acc1 = acc0, acc2 = acc0, acc3 = acc0;

  for (int k0 = 0; k0 < K; k0 += 32) {
    __syncthreads();
    // stage 32x64 W tile as packed f16 pairs; col-fastest => coalesced reads
    for (int i = threadIdx.x; i < 64 * 16; i += GEMM_WAVES * 32) {
      int kp = i >> 6;          // k-pair 0..15
      int cc = i & 63;          // column 0..63
      union { _Float16 h[2]; unsigned int u; } pk;
      pk.h[0] = (_Float16)W[(k0 + 2 * kp)     * Nout + col0 + cc];
      pk.h[1] = (_Float16)W[(k0 + 2 * kp + 1) * Nout + col0 + cc];
      sW2[cc * KPSTR + kp] = pk.u;
    }
    __syncthreads();

    // A fragment (contiguous k order, same as B): lanes 0-15 hold K {0..15},
    // lanes 16-31 hold K {16..31}, row M = lane%16.
    const int ka = k0 + (hi ? 16 : 0);
    const float4 a0 = *(const float4*)(xp + ka);
    const float4 a1 = *(const float4*)(xp + ka + 4);
    const float4 a2 = *(const float4*)(xp + ka + 8);
    const float4 a3 = *(const float4*)(xp + ka + 12);
    v16h a;
    a[0]=(_Float16)a0.x; a[1]=(_Float16)a0.y; a[2]=(_Float16)a0.z; a[3]=(_Float16)a0.w;
    a[4]=(_Float16)a1.x; a[5]=(_Float16)a1.y; a[6]=(_Float16)a1.z; a[7]=(_Float16)a1.w;
    a[8]=(_Float16)a2.x; a[9]=(_Float16)a2.y; a[10]=(_Float16)a2.z; a[11]=(_Float16)a2.w;
    a[12]=(_Float16)a3.x; a[13]=(_Float16)a3.y; a[14]=(_Float16)a3.z; a[15]=(_Float16)a3.w;

    // B fragments: lane reads col {0,16,32,48}+lm, k-half kh (dwords), each
    // fragment = 8 consecutive dwords = two 16B LDS loads (16B-aligned: KPSTR%4==0).
    const int kh = hi ? 8 : 0;
    v16h b0, b1, b2, b3;
    {
      const unsigned int* p0 = &sW2[(lm)      * KPSTR + kh];
      const unsigned int* p1 = &sW2[(lm + 16) * KPSTR + kh];
      const unsigned int* p2 = &sW2[(lm + 32) * KPSTR + kh];
      const unsigned int* p3 = &sW2[(lm + 48) * KPSTR + kh];
      ((uint4*)&b0)[0] = ((const uint4*)p0)[0]; ((uint4*)&b0)[1] = ((const uint4*)p0)[1];
      ((uint4*)&b1)[0] = ((const uint4*)p1)[0]; ((uint4*)&b1)[1] = ((const uint4*)p1)[1];
      ((uint4*)&b2)[0] = ((const uint4*)p2)[0]; ((uint4*)&b2)[1] = ((const uint4*)p2)[1];
      ((uint4*)&b3)[0] = ((const uint4*)p3)[0]; ((uint4*)&b3)[1] = ((const uint4*)p3)[1];
    }

    acc0 = __builtin_amdgcn_wmma_f32_16x16x32_f16(false, a, false, b0, (short)0, acc0, false, false);
    acc1 = __builtin_amdgcn_wmma_f32_16x16x32_f16(false, a, false, b1, (short)0, acc1, false, false);
    acc2 = __builtin_amdgcn_wmma_f32_16x16x32_f16(false, a, false, b2, (short)0, acc2, false, false);
    acc3 = __builtin_amdgcn_wmma_f32_16x16x32_f16(false, a, false, b3, (short)0, acc3, false, false);
  }

  // C/D layout: VGPR i -> row i (lanes 0-15) / row i+8 (lanes 16-31), col lane%16
  const int r0 = mtile * 16 + (hi ? 8 : 0);
  const int cb = col0 + lm;
#pragma unroll
  for (int i = 0; i < 8; ++i) {
    const int ro = (r0 + i) * Nout;
    Y[ro + cb]      = acc0[i];
    Y[ro + cb + 16] = acc1[i];
    Y[ro + cb + 32] = acc2[i];
    Y[ro + cb + 48] = acc3[i];
  }
}

// a_s[n,h] = <xl[n,h,:], att_src[h,:]>, a_d likewise
__global__ void scores_kernel(const float* __restrict__ xl,
                              const float* __restrict__ asrc,
                              const float* __restrict__ adst,
                              float* __restrict__ a_s, float* __restrict__ a_d,
                              int H, int C) {
  int idx = blockIdx.x * blockDim.x + threadIdx.x;
  if (idx >= NNODES * H) return;
  int n = idx / H, h = idx - n * H;
  const float* xp = xl + n * (H * C) + h * C;
  float ss = 0.f, sd = 0.f;
  for (int c = 0; c < C; ++c) {
    float v = xp[c];
    ss += v * asrc[h * C + c];
    sd += v * adst[h * C + c];
  }
  a_s[idx] = ss;
  a_d[idx] = sd;
}

// pass 1: alpha = leaky_relu(a_s[src]+a_d[dst]); segment max into mbuf[dst,h]
__global__ void edge_alpha_max(const int* __restrict__ ei,
                               const float* __restrict__ a_s,
                               const float* __restrict__ a_d,
                               float* __restrict__ alpha,
                               float* __restrict__ mbuf, int H) {
  int idx = blockIdx.x * blockDim.x + threadIdx.x;
  if (idx >= ETOT * H) return;
  int e = idx / H, h = idx - e * H;
  int s = (e < NEDGES) ? ei[e]          : (e - NEDGES);
  int d = (e < NEDGES) ? ei[NEDGES + e] : (e - NEDGES);
  float v = a_s[s * H + h] + a_d[d * H + h];
  v = (v > 0.0f) ? v : 0.2f * v;          // leaky_relu, slope 0.2
  alpha[idx] = v;
  atomicMaxF(&mbuf[d * H + h], v);
}

// pass 2: e = exp(alpha - m[dst]); segment sum into sbuf[dst,h]
__global__ void edge_exp_sum(const int* __restrict__ ei,
                             float* __restrict__ alpha,
                             const float* __restrict__ mbuf,
                             float* __restrict__ sbuf, int H) {
  int idx = blockIdx.x * blockDim.x + threadIdx.x;
  if (idx >= ETOT * H) return;
  int e = idx / H, h = idx - e * H;
  int d = (e < NEDGES) ? ei[NEDGES + e] : (e - NEDGES);
  float m = mbuf[d * H + h];
  if (!isfinite(m)) m = 0.0f;             // reference's isfinite guard
  float v = expf(alpha[idx] - m);
  alpha[idx] = v;
  atomicAdd(&sbuf[d * H + h], v);
}

// pass 3: agg[dst] += (e / s[dst]) * xl[src]   (float4 groups, atomic f32 add)
__global__ void edge_aggregate(const int* __restrict__ ei,
                               const float* __restrict__ alpha,
                               const float* __restrict__ sbuf,
                               const float* __restrict__ xl,
                               float* __restrict__ agg, int H, int C) {
  const int HC = H * C;
  const int groups = HC >> 2;
  int idx = blockIdx.x * blockDim.x + threadIdx.x;
  if (idx >= ETOT * groups) return;       // <= 54.4M, fits int
  int e = idx / groups, j = idx - e * groups;
  int s = (e < NEDGES) ? ei[e]          : (e - NEDGES);
  int d = (e < NEDGES) ? ei[NEDGES + e] : (e - NEDGES);
  int h = (j << 2) / C;
  float coef = alpha[e * H + h] / (sbuf[d * H + h] + 1e-16f);
  const float4 xv = *(const float4*)(xl + s * HC + (j << 2));
  float* op = agg + d * HC + (j << 2);
  atomicAdd(op + 0, coef * xv.x);
  atomicAdd(op + 1, coef * xv.y);
  atomicAdd(op + 2, coef * xv.z);
  atomicAdd(op + 3, coef * xv.w);
}

// out = elu(in + bias)  (concat layers)
__global__ void bias_elu(const float* __restrict__ in, const float* __restrict__ b,
                         float* __restrict__ outp, int HC) {
  int idx = blockIdx.x * blockDim.x + threadIdx.x;
  if (idx >= NNODES * HC) return;
  float v = in[idx] + b[idx % HC];
  outp[idx] = (v > 0.0f) ? v : (expf(v) - 1.0f);
}

// final: out[n,:2] = (g[n]+b3) @ fc_w + fc_b   (layer-3 bias fused here)
__global__ void fc_kernel(const float* __restrict__ g, const float* __restrict__ b3,
                          const float* __restrict__ fcw, const float* __restrict__ fcb,
                          float* __restrict__ outp) {
  int n = blockIdx.x * blockDim.x + threadIdx.x;
  if (n >= NNODES) return;
  float acc0 = fcb[0], acc1 = fcb[1];
  const float* gp = g + n * HID;
#pragma unroll 8
  for (int c = 0; c < HID; ++c) {
    float v = gp[c] + b3[c];
    acc0 += v * fcw[c * 2 + 0];
    acc1 += v * fcw[c * 2 + 1];
  }
  outp[n * 2 + 0] = acc0;
  outp[n * 2 + 1] = acc1;
}

// ---------------------------------------------------------------------------
static void run_gat_layer(const float* xin, int Fin, const float* Wl,
                          const float* asrc, const float* adst, const int* ei,
                          int H, int C, float* xl, float* agg, float* alpha,
                          float* a_s, float* a_d, float* mbuf, float* sbuf,
                          hipStream_t stream) {
  const int HC = H * C;
  dim3 gg(NNODES / (16 * GEMM_WAVES), HC / 64);
  gemm_wmma<<<gg, GEMM_WAVES * 32, 0, stream>>>(xin, Wl, xl, Fin, HC);

  const int nh = NNODES * H;
  scores_kernel<<<(nh + 255) / 256, 256, 0, stream>>>(xl, asrc, adst, a_s, a_d, H, C);
  fill_kernel<<<(nh + 255) / 256, 256, 0, stream>>>(mbuf, -INFINITY, nh);
  fill_kernel<<<(nh + 255) / 256, 256, 0, stream>>>(sbuf, 0.0f, nh);
  const int nagg = NNODES * HC;
  fill_kernel<<<(nagg + 255) / 256, 256, 0, stream>>>(agg, 0.0f, nagg);

  const int eh = ETOT * H;
  edge_alpha_max<<<(eh + 255) / 256, 256, 0, stream>>>(ei, a_s, a_d, alpha, mbuf, H);
  edge_exp_sum <<<(eh + 255) / 256, 256, 0, stream>>>(ei, alpha, mbuf, sbuf, H);
  const int na = ETOT * (HC >> 2);
  edge_aggregate<<<(na + 255) / 256, 256, 0, stream>>>(ei, alpha, sbuf, xl, agg, H, C);
}

extern "C" void kernel_launch(void* const* d_in, const int* in_sizes, int n_in,
                              void* d_out, int out_size, void* d_ws, size_t ws_size,
                              hipStream_t stream) {
  const float* x   = (const float*)d_in[0];
  const int*   ei  = (const int*)d_in[1];     // jnp int64 -> int32 under default JAX x32
  const float* W1  = (const float*)d_in[2];
  const float* as1 = (const float*)d_in[3];
  const float* ad1 = (const float*)d_in[4];
  const float* b1  = (const float*)d_in[5];
  const float* W2  = (const float*)d_in[6];
  const float* as2 = (const float*)d_in[7];
  const float* ad2 = (const float*)d_in[8];
  const float* b2  = (const float*)d_in[9];
  const float* W3  = (const float*)d_in[10];
  const float* as3 = (const float*)d_in[11];
  const float* ad3 = (const float*)d_in[12];
  const float* b3  = (const float*)d_in[13];
  const float* fcw = (const float*)d_in[14];
  const float* fcb = (const float*)d_in[15];
  float* out = (float*)d_out;

  // workspace layout (floats); total ~42.6M floats (~170 MB)
  float* ws = (float*)d_ws;
  size_t off = 0;
  float* xl    = ws + off; off += (size_t)NNODES * 256;
  float* agg   = ws + off; off += (size_t)NNODES * 256;
  float* hbuf  = ws + off; off += (size_t)NNODES * 256;
  float* alpha = ws + off; off += (size_t)ETOT * HEADS;
  float* a_s   = ws + off; off += (size_t)NNODES * HEADS;
  float* a_d   = ws + off; off += (size_t)NNODES * HEADS;
  float* mbuf  = ws + off; off += (size_t)NNODES * HEADS;
  float* sbuf  = ws + off; off += (size_t)NNODES * HEADS;
  (void)in_sizes; (void)n_in; (void)out_size; (void)ws_size;

  // Layer 1: [N,128] -> [N,4,64], concat -> elu
  run_gat_layer(x, INDIM, W1, as1, ad1, ei, HEADS, HID,
                xl, agg, alpha, a_s, a_d, mbuf, sbuf, stream);
  bias_elu<<<(NNODES * 256 + 255) / 256, 256, 0, stream>>>(agg, b1, hbuf, 256);

  // Layer 2: [N,256] -> [N,4,64], concat -> elu
  run_gat_layer(hbuf, HEADS * HID, W2, as2, ad2, ei, HEADS, HID,
                xl, agg, alpha, a_s, a_d, mbuf, sbuf, stream);
  bias_elu<<<(NNODES * 256 + 255) / 256, 256, 0, stream>>>(agg, b2, hbuf, 256);

  // Layer 3: [N,256] -> [N,1,64], mean over 1 head == identity; bias fused in fc
  run_gat_layer(hbuf, HEADS * HID, W3, as3, ad3, ei, 1, HID,
                xl, agg, alpha, a_s, a_d, mbuf, sbuf, stream);

  fc_kernel<<<(NNODES + 255) / 256, 256, 0, stream>>>(agg, b3, fcw, fcb, out);
}
